// Effi_MVS_51659866636482
// MI455X (gfx1250) — matrix operand order repaired and checked by reading, not compiled, not run on hardware
//
#include <hip/hip_runtime.h>
#include <hip/hip_bf16.h>

// ---------------------------------------------------------------------------
// Problem constants
// ---------------------------------------------------------------------------
#define NV    5
#define NC    32
#define ND    48
#define NH    128
#define NW    160
#define RATIO 8
#define HW    (NH * NW)          // 20480
#define DHW   (ND * NH * NW)     // 983040

// Padded variance volume (zero halo of 1 voxel in d/h/w), channel-last
#define DP    (ND + 2)           // 50
#define HP    (NH + 2)           // 130
#define WP    (NW + 2)           // 162
#define PVOX  (DP * HP * WP)     // 1,053,000 voxels, 64B each

#define GD    4                  // depths per wave in the WMMA conv

typedef __attribute__((ext_vector_type(16))) __bf16 bf16x16;
typedef __attribute__((ext_vector_type(8)))  float  f32x8;

union BV32 { uint4 u[2]; bf16x16 v; };   // 32 bytes = one WMMA bf16 operand per lane

__device__ __forceinline__ unsigned short f2bf(float f) {
    unsigned int u = __float_as_uint(f);
    unsigned int r = u + 0x7FFFu + ((u >> 16) & 1u);   // round-to-nearest-even
    return (unsigned short)(r >> 16);
}

// ---------------------------------------------------------------------------
// Kernel 0: projection algebra (1 thread) + packed WMMA A-table (32 threads)
// ---------------------------------------------------------------------------
__device__ void combine_mat(const float* p, float* M) {
    // p: [2][4][4] for one view; M = [[K(3x3) @ E(3x4)], E_row3]
    const float* E = p;
    const float* K = p + 16;
    for (int r = 0; r < 3; ++r)
        for (int c = 0; c < 4; ++c) {
            float s = 0.f;
            for (int j = 0; j < 3; ++j) s += K[r * 4 + j] * E[j * 4 + c];
            M[r * 4 + c] = s;
        }
    for (int c = 0; c < 4; ++c) M[12 + c] = E[12 + c];
}

__device__ void mat4_inv(const float* A, float* out) {
    float a[4][8];
    for (int r = 0; r < 4; ++r)
        for (int c = 0; c < 4; ++c) {
            a[r][c] = A[r * 4 + c];
            a[r][4 + c] = (r == c) ? 1.f : 0.f;
        }
    for (int col = 0; col < 4; ++col) {
        int piv = col;
        for (int r = col + 1; r < 4; ++r)
            if (fabsf(a[r][col]) > fabsf(a[piv][col])) piv = r;
        if (piv != col)
            for (int c = 0; c < 8; ++c) { float t = a[col][c]; a[col][c] = a[piv][c]; a[piv][c] = t; }
        float invd = 1.f / a[col][col];
        for (int c = 0; c < 8; ++c) a[col][c] *= invd;
        for (int r = 0; r < 4; ++r) {
            if (r == col) continue;
            float f = a[r][col];
            for (int c = 0; c < 8; ++c) a[r][c] -= f * a[col][c];
        }
    }
    for (int r = 0; r < 4; ++r)
        for (int c = 0; c < 4; ++c) out[r * 4 + c] = a[r][4 + c];
}

__global__ void prep_kernel(const float* __restrict__ proj_mats,  // (1,V,2,4,4)
                            const float* __restrict__ wreg,       // (1,C,3,3,3)
                            float* __restrict__ proj_ws,          // 4 views * (9 rot + 3 trans)
                            unsigned short* __restrict__ atab)    // 27 taps * 32 lanes * 16 bf16
{
    const int lane = threadIdx.x;
    if (lane == 0) {
        float Mref[16], invRef[16];
        combine_mat(proj_mats, Mref);                 // view 0
        mat4_inv(Mref, invRef);
        for (int v = 1; v < NV; ++v) {
            float Mv[16], P[16];
            combine_mat(proj_mats + v * 32, Mv);
            for (int r = 0; r < 4; ++r)
                for (int c = 0; c < 4; ++c) {
                    float s = 0.f;
                    for (int j = 0; j < 4; ++j) s += Mv[r * 4 + j] * invRef[j * 4 + c];
                    P[r * 4 + c] = s;
                }
            float* o = proj_ws + (v - 1) * 12;
            for (int r = 0; r < 3; ++r)
                for (int c = 0; c < 3; ++c) o[r * 3 + c] = P[r * 4 + c];
            o[9] = P[3]; o[10] = P[7]; o[11] = P[11];
        }
    }
    // A-table: row M=0 carries the weight vector over K=C; contiguous k = 16h + s layout (same as B):
    //   lane 0  -> elems 0..7 = K0..7,   elems 8..15 = K8..15
    //   lane 16 -> elems 0..7 = K16..23, elems 8..15 = K24..31
    //   all other lanes (M=1..15) = 0
    for (int t = 0; t < 27; ++t) {
        unsigned short e[16];
        for (int i = 0; i < 16; ++i) e[i] = 0;
        if (lane == 0) {
            for (int i = 0; i < 8; ++i) e[i]     = f2bf(wreg[(i)      * 27 + t]);
            for (int i = 0; i < 8; ++i) e[8 + i] = f2bf(wreg[(8 + i)  * 27 + t]);
        } else if (lane == 16) {
            for (int i = 0; i < 8; ++i) e[i]     = f2bf(wreg[(16 + i) * 27 + t]);
            for (int i = 0; i < 8; ++i) e[8 + i] = f2bf(wreg[(24 + i) * 27 + t]);
        }
        unsigned short* dst = atab + (t * 32 + lane) * 16;
        for (int i = 0; i < 16; ++i) dst[i] = e[i];
    }
}

// ---------------------------------------------------------------------------
// Kernel 0b: zero the halo voxels of the padded variance volume (64B each)
// ---------------------------------------------------------------------------
__global__ __launch_bounds__(256) void halo_zero_kernel(unsigned short* __restrict__ var)
{
    const int idx = blockIdx.x * 256 + threadIdx.x;
    if (idx >= PVOX) return;
    const int wp = idx % WP;
    const int t  = idx / WP;
    const int hp = t % HP;
    const int dp = t / HP;
    if (dp == 0 || dp == DP - 1 || hp == 0 || hp == HP - 1 || wp == 0 || wp == WP - 1) {
        uint4* dst = (uint4*)(var + (size_t)idx * NC);
        const uint4 z = make_uint4(0u, 0u, 0u, 0u);
        dst[0] = z; dst[1] = z; dst[2] = z; dst[3] = z;
    }
}

// ---------------------------------------------------------------------------
// Kernel 1: homography warp + running variance, store bf16 channel-last into
// the INTERIOR of the padded volume: voxel (d,h,w) -> padded (d+1,h+1,w+1)
// ---------------------------------------------------------------------------
__global__ __launch_bounds__(256) void warp_var_kernel(
    const float* __restrict__ feat,          // (V,1,C,H,W)
    const float* __restrict__ proj_ws,       // 4 * 12
    const float* __restrict__ depth_values,  // (1,D)
    unsigned short* __restrict__ var)        // PVOX * 32 bf16 (padded)
{
    const int idx = blockIdx.x * 256 + threadIdx.x;   // exactly DHW threads
    const int w = idx % NW;
    const int t = idx / NW;
    const int h = t % NH;
    const int d = t / NH;
    const float depth = depth_values[d];

    float s[NC], q[NC];
    {
        const float* f0 = feat + h * NW + w;          // view 0, no warp
        #pragma unroll
        for (int c = 0; c < NC; ++c) {
            float v = f0[c * HW];
            s[c] = v; q[c] = v * v;
        }
    }

    const float fx = (float)w, fy = (float)h;
    for (int vi = 0; vi < NV - 1; ++vi) {
        const float* pr = proj_ws + vi * 12;
        float rx = pr[0] * fx + pr[1] * fy + pr[2];
        float ry = pr[3] * fx + pr[4] * fy + pr[5];
        float rz = pr[6] * fx + pr[7] * fy + pr[8];
        float px = rx * depth + pr[9];
        float py = ry * depth + pr[10];
        float pz = rz * depth + pr[11];
        float gx = px / pz, gy = py / pz;
        float x0 = floorf(gx), y0 = floorf(gy);
        float wx = gx - x0, wy = gy - y0;
        float x1 = x0 + 1.f, y1 = y0 + 1.f;
        float vx0 = (x0 >= 0.f && x0 <= (float)(NW - 1)) ? 1.f : 0.f;
        float vx1 = (x1 >= 0.f && x1 <= (float)(NW - 1)) ? 1.f : 0.f;
        float vy0 = (y0 >= 0.f && y0 <= (float)(NH - 1)) ? 1.f : 0.f;
        float vy1 = (y1 >= 0.f && y1 <= (float)(NH - 1)) ? 1.f : 0.f;
        float w00 = (1.f - wx) * (1.f - wy) * vx0 * vy0;
        float w10 = wx * (1.f - wy) * vx1 * vy0;
        float w01 = (1.f - wx) * wy * vx0 * vy1;
        float w11 = wx * wy * vx1 * vy1;
        int xi0 = (int)fminf(fmaxf(x0, 0.f), (float)(NW - 1));
        int xi1 = (int)fminf(fmaxf(x1, 0.f), (float)(NW - 1));
        int yi0 = (int)fminf(fmaxf(y0, 0.f), (float)(NH - 1));
        int yi1 = (int)fminf(fmaxf(y1, 0.f), (float)(NH - 1));
        int i00 = yi0 * NW + xi0, i10 = yi0 * NW + xi1;
        int i01 = yi1 * NW + xi0, i11 = yi1 * NW + xi1;
        const float* fb = feat + (vi + 1) * NC * HW;
        #pragma unroll
        for (int c = 0; c < NC; ++c) {
            const float* fc = fb + c * HW;
            float val = w00 * fc[i00] + w10 * fc[i10] + w01 * fc[i01] + w11 * fc[i11];
            s[c] += val; q[c] += val * val;
        }
    }

    union { unsigned short us[NC]; uint4 u4[4]; } pack;
    #pragma unroll
    for (int c = 0; c < NC; ++c) {
        float m = s[c] * (1.f / NV);
        pack.us[c] = f2bf(q[c] * (1.f / NV) - m * m);
    }
    const size_t pvox = (size_t)(((d + 1) * HP + (h + 1)) * WP + (w + 1));
    uint4* dst = (uint4*)(var + pvox * NC);
    dst[0] = pack.u4[0]; dst[1] = pack.u4[1]; dst[2] = pack.u4[2]; dst[3] = pack.u4[3];
}

// ---------------------------------------------------------------------------
// Kernel 2: 3x3x3 x C=32 -> 1 conv via bf16 WMMA.
//   One wave computes GD=4 consecutive depth outputs at one (h, 16-w group).
//   Per (dy,dx): load the 3 dz A-taps once, then stream 6 B planes (k=0..5);
//   each B plane feeds up to 3 accumulators: acc[k-dz] += A[dz] x B_k.
//   108 WMMAs / 54 B-loads / 27 A-loads per wave; halo padding -> no
//   boundary divergence in the hot loop.
// ---------------------------------------------------------------------------
__global__ __launch_bounds__(256) void conv_wmma_kernel(
    const unsigned short* __restrict__ var,   // padded [dp][hp][wp][c] bf16
    const unsigned short* __restrict__ atab,  // [tap][lane][16] bf16
    float* __restrict__ pre)                  // [d][h][w] f32
{
    const int lane  = threadIdx.x & 31;
    const int wave  = threadIdx.x >> 5;
    const int group = blockIdx.x * 8 + wave;  // exactly (ND/GD)*NH*(NW/16) = 15360
    const int wgrp = group % (NW / 16);
    const int hd   = group / (NW / 16);
    const int h0   = hd % NH;
    const int d0   = (hd / NH) * GD;
    const int w0   = wgrp * 16;
    const int n     = lane & 15;   // N (pixel) index
    const int khalf = lane >> 4;   // 0: K=0..15, 1: K=16..31 (contiguous k order, as A-table)

    f32x8 acc[GD];
    #pragma unroll
    for (int g = 0; g < GD; ++g)
        acc[g] = (f32x8){0.f, 0.f, 0.f, 0.f, 0.f, 0.f, 0.f, 0.f};

    // Per-lane channel-half offset (in ushorts) within a voxel
    const size_t kofs = (size_t)khalf * 16;

    #pragma unroll
    for (int dy = 0; dy < 3; ++dy) {
        #pragma unroll
        for (int dx = 0; dx < 3; ++dx) {
            // A operands for dz = 0,1,2 at this (dy,dx)
            BV32 a[3];
            #pragma unroll
            for (int dz = 0; dz < 3; ++dz) {
                const int tp = dz * 9 + dy * 3 + dx;
                const uint4* ap = (const uint4*)(atab + (tp * 32 + lane) * 16);
                a[dz].u[0] = ap[0];
                a[dz].u[1] = ap[1];
            }
            // Padded spatial position for this (dy,dx): hp = h0+dy, wp = w0+n+dx
            const size_t base =
                ((size_t)((d0 * HP + (h0 + dy)) * WP + (w0 + n + dx))) * NC + kofs;
            #pragma unroll
            for (int k = 0; k < GD + 2; ++k) {           // 6 padded d-planes
                BV32 b;
                const uint4* bp =
                    (const uint4*)(var + base + (size_t)k * (size_t)(HP * WP) * NC);
                b.u[0] = bp[0];
                b.u[1] = bp[1];
                #pragma unroll
                for (int dz = 0; dz < 3; ++dz) {
                    const int g = k - dz;
                    if (g >= 0 && g < GD) {
                        acc[g] = __builtin_amdgcn_wmma_f32_16x16x32_bf16(
                            /*neg_a=*/false, a[dz].v, /*neg_b=*/false, b.v,
                            /*c_mod=*/(short)0, acc[g],
                            /*reuse_a=*/false, /*reuse_b=*/false);
                    }
                }
            }
        }
    }

    // D row M=0 lives in VGPR0 on lanes 0..15 (N = lane)
    if (lane < 16) {
        #pragma unroll
        for (int g = 0; g < GD; ++g)
            pre[((d0 + g) * NH + h0) * NW + w0 + lane] = acc[g][0];
    }
}

// ---------------------------------------------------------------------------
// Kernel 3: softmax over D, expected depth, confidence
// ---------------------------------------------------------------------------
__global__ __launch_bounds__(256) void softmax_depth_kernel(
    const float* __restrict__ pre, const float* __restrict__ depth_values,
    float* __restrict__ depth_ws, float* __restrict__ out_depth,
    float* __restrict__ out_conf)
{
    const int idx = blockIdx.x * 256 + threadIdx.x;   // exactly HW threads
    float m = -3.402823466e38f;
    for (int d = 0; d < ND; ++d) m = fmaxf(m, pre[d * HW + idx]);
    float ssum = 0.f;
    for (int d = 0; d < ND; ++d) ssum += __expf(pre[d * HW + idx] - m);
    const float inv = 1.f / ssum;
    float dep = 0.f, didx = 0.f;
    for (int d = 0; d < ND; ++d) {
        float p = __expf(pre[d * HW + idx] - m) * inv;
        dep  += p * depth_values[d];
        didx += p * (float)d;
    }
    int di = (int)didx;
    di = min(max(di, 0), ND - 1);
    float conf = 0.f;
    for (int dd = di - 1; dd <= di + 2; ++dd)
        if (dd >= 0 && dd < ND) conf += __expf(pre[dd * HW + idx] - m) * inv;
    depth_ws[idx] = dep;
    out_depth[idx] = dep;
    out_conf[idx]  = conf;
}

// ---------------------------------------------------------------------------
// Kernel 4: convex upsample (softmax over 9 weights, 3x3 depth neighborhood)
// ---------------------------------------------------------------------------
__global__ __launch_bounds__(256) void upsample_kernel(
    const float* __restrict__ mask,      // (1, 9*64, H, W)
    const float* __restrict__ depth_ws,  // (H, W)
    float* __restrict__ out)             // (r*H, r*W)
{
    const int idx = blockIdx.x * 256 + threadIdx.x;   // exactly r*H*r*W threads
    const int x = idx % (NW * RATIO);
    const int y = idx / (NW * RATIO);
    const int w = x >> 3, j = x & 7;
    const int h = y >> 3, i = y & 7;

    float mv[9];
    float mmax = -3.402823466e38f;
    #pragma unroll
    for (int k = 0; k < 9; ++k) {
        mv[k] = mask[((k * RATIO + i) * RATIO + j) * HW + h * NW + w];
        mmax = fmaxf(mmax, mv[k]);
    }
    float ssum = 0.f;
    #pragma unroll
    for (int k = 0; k < 9; ++k) { mv[k] = __expf(mv[k] - mmax); ssum += mv[k]; }
    const float inv = 1.f / ssum;
    float acc = 0.f;
    #pragma unroll
    for (int k = 0; k < 9; ++k) {
        const int hh = h + k / 3 - 1;
        const int ww = w + k % 3 - 1;
        float dnb = ((unsigned)hh < (unsigned)NH && (unsigned)ww < (unsigned)NW)
                        ? depth_ws[hh * NW + ww] : 0.f;
        acc += mv[k] * inv * dnb;
    }
    out[idx] = acc;
}

// ---------------------------------------------------------------------------
// Launch
// ---------------------------------------------------------------------------
extern "C" void kernel_launch(void* const* d_in, const int* in_sizes, int n_in,
                              void* d_out, int out_size, void* d_ws, size_t ws_size,
                              hipStream_t stream) {
    const float* features     = (const float*)d_in[0];
    const float* proj_mats    = (const float*)d_in[1];
    const float* depth_values = (const float*)d_in[2];
    const float* mask         = (const float*)d_in[3];
    const float* wreg         = (const float*)d_in[4];
    float* out = (float*)d_out;

    char* ws = (char*)d_ws;
    const size_t VAR_BYTES = (size_t)PVOX * NC * 2;               // 67,392,000
    float*          proj_ws = (float*)ws;                          // 192 B
    unsigned short* atab    = (unsigned short*)(ws + 1024);        // 27,648 B
    unsigned short* var     = (unsigned short*)(ws + 65536);       // padded bf16 volume
    float*          pre     = (float*)(ws + 65536 + VAR_BYTES);    // 3,932,160 B
    float*          dep_ws  = (float*)(ws + 65536 + VAR_BYTES + (size_t)DHW * 4);

    float* out_depth_up = out;                       // 1,310,720
    float* out_depth    = out + (size_t)HW * RATIO * RATIO;  // 20,480 @ 1,310,720
    float* out_conf     = out_depth + HW;            // 20,480

    prep_kernel<<<1, 32, 0, stream>>>(proj_mats, wreg, proj_ws, atab);
    halo_zero_kernel<<<(PVOX + 255) / 256, 256, 0, stream>>>(var);
    warp_var_kernel<<<DHW / 256, 256, 0, stream>>>(features, proj_ws, depth_values, var);
    conv_wmma_kernel<<<((ND / GD) * NH * (NW / 16)) / 8, 256, 0, stream>>>(var, atab, pre);
    softmax_depth_kernel<<<HW / 256, 256, 0, stream>>>(pre, depth_values, dep_ws,
                                                       out_depth, out_conf);
    upsample_kernel<<<(HW * RATIO * RATIO) / 256, 256, 0, stream>>>(mask, dep_ws,
                                                                    out_depth_up);
}
